// BinaryTreeLSTM_65017214926869
// MI455X (gfx1250) — matrix operand order repaired and checked by reading, not compiled, not run on hardware
//
#include <hip/hip_runtime.h>
#include <hip/hip_bf16.h>

#define S_LEN 128
#define BATCH 512
#define DIM   256
#define KTOT  768        // [x | lh | rh] along K
#define NTOT  1280       // 5 gates * DIM
#define NT_TILES 80      // NTOT/16
#define KT_TILES 24      // KTOT/32
#define LDA   776        // padded LDS row stride for A (bf16 elements)
#define FRAG_ELEMS 512   // one 16x32 bf16 B fragment = 32 lanes * 16 elems

typedef __attribute__((ext_vector_type(16))) __bf16 v16bf;
typedef __attribute__((ext_vector_type(8)))  __bf16 v8bf;
typedef __attribute__((ext_vector_type(8)))  float  v8f;

__device__ __forceinline__ float sigf(float x) {
    return 1.0f / (1.0f + __expf(-x));
}

// ---------------------------------------------------------------------------
// Prep 1: recover gather indices from one-hot masks. tid < 2*S*B, first half
// handles left mask, second half right mask. Index = -1 means "row is zero".
// ---------------------------------------------------------------------------
__global__ void prep_indices(const float* __restrict__ lmask,
                             const float* __restrict__ rmask,
                             int* __restrict__ li, int* __restrict__ ri) {
    int tid = blockIdx.x * blockDim.x + threadIdx.x;
    if (tid >= 2 * S_LEN * BATCH) return;
    int which = tid / (S_LEN * BATCH);
    int sb = tid - which * (S_LEN * BATCH);
    const float* row = (which == 0 ? lmask : rmask) + (size_t)sb * S_LEN;
    int idx = -1;
    for (int t = 0; t < S_LEN; ++t)
        if (row[t] > 0.5f) idx = t;
    if (which == 0) li[sb] = idx; else ri[sb] = idx;
}

// ---------------------------------------------------------------------------
// Prep 2: pack combined weight matrix Wcomb (KTOT x NTOT) into bf16 WMMA
// B-fragment order: Wpack[((nt*KT_TILES + kt)*32 + lane)*16 + e]
// B 32x16 bf16 layout: lane = (N & 15) + 16*Khalf ; element e<8 -> K = Khalf*8+e, e>=8 -> K = 16+Khalf*8+(e-8)
// Column n = (gate g, out k): K<256 -> Wx[gx(g)][k][K], K<512 -> Wlh[g][k][K-256],
// else Wrh[g][k][K-512].   gx = {0,1,2,2,3}.
// ---------------------------------------------------------------------------
__global__ void prep_wpack(const float* __restrict__ Wx,
                           const float* __restrict__ Wlh,
                           const float* __restrict__ Wrh,
                           __bf16* __restrict__ Wpack) {
    int tid = blockIdx.x * blockDim.x + threadIdx.x;
    if (tid >= NT_TILES * KT_TILES * FRAG_ELEMS) return;
    int e    = tid & 15;
    int lane = (tid >> 4) & 31;
    int t2   = tid >> 9;
    int kt   = t2 % KT_TILES;
    int nt   = t2 / KT_TILES;

    int n  = lane & 15;
    int hh = lane >> 4;
    int kk = kt * 32 + ((e < 8) ? (hh * 8 + e) : (16 + hh * 8 + (e - 8)));     // K index in [0, KTOT)
    int nglob = nt * 16 + n;            // N index in [0, NTOT)
    int g = nglob >> 8;                 // gate 0..4
    int k = nglob & 255;                // output feature
    int gxg = (g <= 1) ? g : ((g <= 3) ? 2 : 3);

    float w;
    if (kk < DIM) {
        w = Wx[((size_t)gxg * DIM + k) * DIM + kk];
    } else if (kk < 2 * DIM) {
        w = Wlh[((size_t)g * DIM + k) * DIM + (kk - DIM)];
    } else {
        w = Wrh[((size_t)g * DIM + k) * DIM + (kk - 2 * DIM)];
    }
    Wpack[tid] = (__bf16)w;
}

// ---------------------------------------------------------------------------
// One scan step. Grid = BATCH/16 workgroups, 256 threads (8 waves).
// Each WG: 16 batch rows. Stage A=[x[s]|lh|rh] (bf16) in LDS, GEMM vs Wpack
// with v_wmma_f32_16x16x32_bf16, dump P (16x1280 f32) to LDS, fused epilogue.
// All B-fragment / A-fragment addresses are hoisted to one base pointer so
// the unrolled loads fold into immediate offsets.
// ---------------------------------------------------------------------------
__global__ __launch_bounds__(256) void tree_lstm_step(
    int s,
    const float* __restrict__ x,       // (S,B,D)
    const float* __restrict__ x_mask,  // (S,B)
    const float* __restrict__ bx,      // (4,D)
    const int*   __restrict__ li_idx,  // (S,B)
    const int*   __restrict__ ri_idx,  // (S,B)
    const __bf16* __restrict__ Wpack,  // fragment-packed weights
    float* __restrict__ h,             // (B,S,D) = d_out
    float* __restrict__ c)             // (B,S,D) workspace
{
    extern __shared__ char smem[];
    float*  P = (float*)smem;                        // 16 * NTOT f32
    __bf16* A = (__bf16*)(smem + 16 * NTOT * 4);     // 16 * LDA bf16

    const int tid   = threadIdx.x;
    const int mbase = blockIdx.x * 16;

    // ---- stage A tile (16 x 768) into LDS as bf16 ----
    for (int idx = tid; idx < 16 * KTOT; idx += 256) {
        int row = idx / KTOT;
        int col = idx - row * KTOT;
        int b = mbase + row;
        float v;
        if (col < DIM) {
            v = x[((size_t)s * BATCH + b) * DIM + col];
        } else if (col < 2 * DIM) {
            int t = li_idx[s * BATCH + b];
            v = (t >= 0) ? h[((size_t)b * S_LEN + t) * DIM + (col - DIM)] : 0.0f;
        } else {
            int t = ri_idx[s * BATCH + b];
            v = (t >= 0) ? h[((size_t)b * S_LEN + t) * DIM + (col - 2 * DIM)] : 0.0f;
        }
        A[row * LDA + col] = (__bf16)v;
    }
    __syncthreads();

    // ---- GEMM: each wave owns 10 N-tiles of 16 columns ----
    const int wave = tid >> 5;
    const int lane = tid & 31;
    const int mrow = lane & 15;
    const int hh   = lane >> 4;

    // Hoisted bases: per-lane A row base in LDS, per-lane B slice base in global.
    const __bf16* __restrict__ abase = &A[mrow * LDA + hh * 8];
    const __bf16* __restrict__ wbase =
        Wpack + ((size_t)wave * 10 * KT_TILES) * FRAG_ELEMS + (size_t)lane * 16;

    v8f acc[10];
    #pragma unroll
    for (int t = 0; t < 10; ++t)
        #pragma unroll
        for (int j = 0; j < 8; ++j) acc[t][j] = 0.0f;

    #pragma unroll 1
    for (int kt = 0; kt < KT_TILES; ++kt) {
        // A fragment 16x32 bf16: lane (m, khalf): elems 0..7 -> K = hh*8+e,
        // elems 8..15 -> K = 16 + hh*8 + e   (two 16B LDS vector loads,
        // kt contributes a constant byte offset -> folded into ds_load offset)
        v8bf alo = *(const v8bf*)(abase + kt * 32);
        v8bf ahi = *(const v8bf*)(abase + kt * 32 + 16);
        v16bf afrag;
        #pragma unroll
        for (int e = 0; e < 8; ++e) { afrag[e] = alo[e]; afrag[e + 8] = ahi[e]; }

        const __bf16* __restrict__ wkt = wbase + kt * FRAG_ELEMS;
        #pragma unroll
        for (int t = 0; t < 10; ++t) {
            // constant offset t*KT_TILES*FRAG_ELEMS -> immediate in global_load
            v16bf bfrag = *(const v16bf*)(wkt + t * (KT_TILES * FRAG_ELEMS));
            acc[t] = __builtin_amdgcn_wmma_f32_16x16x32_bf16(
                false, afrag, false, bfrag, (short)0, acc[t], false, false);
        }
    }

    // ---- scatter accumulators (C/D layout: lane=N, vgpr j -> M = 8*hh + j) ----
    #pragma unroll
    for (int t = 0; t < 10; ++t) {
        int ncol = (wave * 10 + t) * 16 + (lane & 15);
        #pragma unroll
        for (int j = 0; j < 8; ++j)
            P[(8 * hh + j) * NTOT + ncol] = acc[t][j];
    }
    __syncthreads();

    // ---- fused epilogue: gates, c gathers, masked row-s write ----
    for (int idx = tid; idx < 16 * DIM; idx += 256) {
        int row = idx >> 8;
        int k   = idx & 255;
        int b = mbase + row;
        const float* pr = &P[row * NTOT];
        float u  = tanhf(pr[0 * DIM + k] + bx[0 * DIM + k]);
        float ig = sigf (pr[1 * DIM + k] + bx[1 * DIM + k]);
        float lf = sigf (pr[2 * DIM + k] + bx[2 * DIM + k]);
        float rf = sigf (pr[3 * DIM + k] + bx[2 * DIM + k]);   // shares bx[2]
        float og = sigf (pr[4 * DIM + k] + bx[3 * DIM + k]);

        int lt = li_idx[s * BATCH + b];
        int rt = ri_idx[s * BATCH + b];
        float lc = (lt >= 0) ? c[((size_t)b * S_LEN + lt) * DIM + k] : 0.0f;
        float rc = (rt >= 0) ? c[((size_t)b * S_LEN + rt) * DIM + k] : 0.0f;

        float cn = ig * u + lf * lc + rf * rc;
        float hn = og * tanhf(cn);
        float m  = x_mask[s * BATCH + b];

        size_t o = ((size_t)b * S_LEN + s) * DIM + k;
        h[o] = m * hn;   // row s only ever written here (prev value is 0)
        c[o] = m * cn;
    }
}

extern "C" void kernel_launch(void* const* d_in, const int* in_sizes, int n_in,
                              void* d_out, int out_size, void* d_ws, size_t ws_size,
                              hipStream_t stream) {
    const float* x      = (const float*)d_in[0];
    const float* x_mask = (const float*)d_in[1];
    const float* lmask  = (const float*)d_in[2];
    const float* rmask  = (const float*)d_in[3];
    const float* Wx     = (const float*)d_in[4];
    const float* bx     = (const float*)d_in[5];
    const float* Wlh    = (const float*)d_in[6];
    const float* Wrh    = (const float*)d_in[7];
    float* h = (float*)d_out;

    char* ws = (char*)d_ws;
    size_t off = 0;
    float* c = (float*)(ws + off);                       // cell state (B,S,D)
    off += (size_t)BATCH * S_LEN * DIM * sizeof(float);  // 64 MB
    __bf16* Wpack = (__bf16*)(ws + off);                 // ~1.97 MB
    off += (size_t)NT_TILES * KT_TILES * FRAG_ELEMS * sizeof(__bf16);
    off = (off + 255) & ~(size_t)255;
    int* li_idx = (int*)(ws + off);
    off += (size_t)S_LEN * BATCH * sizeof(int);
    int* ri_idx = (int*)(ws + off);

    prep_indices<<<(2 * S_LEN * BATCH + 255) / 256, 256, 0, stream>>>(
        lmask, rmask, li_idx, ri_idx);
    prep_wpack<<<(NT_TILES * KT_TILES * FRAG_ELEMS + 255) / 256, 256, 0, stream>>>(
        Wx, Wlh, Wrh, Wpack);

    size_t lds_bytes = (size_t)16 * NTOT * 4 + (size_t)16 * LDA * 2;
    for (int s = 0; s < S_LEN; ++s) {
        tree_lstm_step<<<BATCH / 16, 256, lds_bytes, stream>>>(
            s, x, x_mask, bx, li_idx, ri_idx, Wpack, h, c);
    }
}
